// RebuiltDatacubeCNN_34900904247651
// MI455X (gfx1250) — matrix operand order repaired and checked by reading, not compiled, not run on hardware
//
#include <hip/hip_runtime.h>
#include <hip/hip_bf16.h>
#include <math.h>

// ---------------------------------------------------------------------------
// CDNA5 (gfx1250) implementation of the 3D U-Net-ish datacube CNN.
// Convs / transposed convs are implicit GEMMs on v_wmma_f32_16x16x32_f16
// (f16 inputs, f32 accumulate). Per block: M=128 (8 waves x 16 px),
// N=32 (two 16-wide Cout tiles, two accumulators per wave), K chunked by 32
// with double-buffered packed-f16 weight staging in LDS. Gathers are
// branchless (clamped unsigned offset + select-zero) so loads pipeline;
// f32->f16x2 packing uses v_cvt_pk_rtz_f16_f32.
// ---------------------------------------------------------------------------

typedef __attribute__((ext_vector_type(16))) _Float16 v16h;
typedef __attribute__((ext_vector_type(2)))  __fp16   v2fp16;
typedef __attribute__((ext_vector_type(8)))  float    v8f;
typedef __attribute__((ext_vector_type(8)))  unsigned v8u;

#define GN_GROUPS 8

static __device__ __forceinline__ unsigned pack_f16x2(float a, float b) {
  v2fp16 h = __builtin_amdgcn_cvt_pkrtz(a, b);   // v_cvt_pk_rtz_f16_f32
  return __builtin_bit_cast(unsigned, h);
}

// ===========================================================================
// conv3d, k=3, pad=1, stride s (1|2), NCDHW / OIDHW, implicit GEMM + WMMA.
// ===========================================================================
__global__ __launch_bounds__(256) void conv3d_k3_wmma(
    const float* __restrict__ x, const float* __restrict__ w,
    const float* __restrict__ bias, float* __restrict__ y,
    int N, int Cin, int Din, int Hin, int Win,
    int Cout, int Dout, int Hout, int Wout, int stride)
{
  // [buf][coTile][kPair][n] ; each entry packs f16 K=2p (lo) and K=2p+1 (hi)
  __shared__ unsigned ldsBp[2][2][16][16];

  const int HWi  = Hin * Win;
  const int HWo  = Hout * Wout;
  const int Mtot = N * Dout * HWo;
  const int Ktot = Cin * 27;

  const int lane  = threadIdx.x & 31;
  const int wave  = threadIdx.x >> 5;
  const int lm    = lane & 15;
  const int lhalf = lane >> 4;

  const int coBase = blockIdx.y * 32;
  const int mBase  = (blockIdx.x * 8 + wave) * 16;
  const int m      = mBase + lm;
  const bool mvalid = (m < Mtot);

  int ow = 0, oh = 0, od = 0, nb = 0;
  {
    int mc = mvalid ? m : 0;
    ow = mc % Wout; int t = mc / Wout;
    oh = t % Hout;  t /= Hout;
    od = t % Dout;  nb = t / Dout;
  }
  const int id0 = od * stride - 1;
  const int ih0 = oh * stride - 1;
  const int iw0 = ow * stride - 1;
  // base offset of (nb, ci=0, id0, ih0, iw0) -- taps added per element
  const int xbase = (nb * Cin) * Din * HWi + id0 * HWi + ih0 * Win + iw0;
  const int ciStride = Din * HWi;

  const int chunks = (Ktot + 31) >> 5;
  const int kbase8 = lhalf * 8;

  // per-element branchless clamped gather (returns f32, 0 if invalid)
  auto gatherA = [&](int kk) -> float {
    int ci = kk / 27, r = kk - ci * 27;
    int kd = r / 9;  r -= kd * 9;
    int kh = r / 3;
    int kw = r - kh * 3;
    int idd = id0 + kd, ihh = ih0 + kh, iww = iw0 + kw;
    bool valid = mvalid && (kk < Ktot) &&
                 ((unsigned)idd < (unsigned)Din) &&
                 ((unsigned)ihh < (unsigned)Hin) &&
                 ((unsigned)iww < (unsigned)Win);
    unsigned off =
        valid ? (unsigned)(xbase + ci * ciStride + kd * HWi + kh * Win + kw) : 0u;
    float v = x[off];
    return valid ? v : 0.f;
  };

  // ---- branchless cooperative weight-chunk stage into LDS ----
  auto loadB = [&](int kc, int buf) {
#pragma unroll
    for (int i = 0; i < 2; ++i) {
      int e    = threadIdx.x + (i << 8);          // 0..511
      int tile = e >> 8, rem = e & 255;
      int kp   = rem >> 4, n = rem & 15;
      int kg   = kc + (kp << 1);
      int co   = coBase + (tile << 4) + n;
      bool cv  = (co < Cout);
      bool v0  = cv && (kg < Ktot);
      bool v1  = cv && (kg + 1 < Ktot);
      int ci0 = kg / 27,       r0 = kg - ci0 * 27;
      int ci1 = (kg + 1) / 27, r1 = (kg + 1) - ci1 * 27;
      unsigned o0 = v0 ? (unsigned)((co * Cin + ci0) * 27 + r0) : 0u;
      unsigned o1 = v1 ? (unsigned)((co * Cin + ci1) * 27 + r1) : 0u;
      float f0 = w[o0], f1 = w[o1];
      f0 = v0 ? f0 : 0.f;
      f1 = v1 ? f1 : 0.f;
      ldsBp[buf][tile][kp][n] = pack_f16x2(f0, f1);
    }
    if (threadIdx.x == 0 && (kc + 32) < Ktot)
      __builtin_prefetch(&w[(coBase * Cin) * 27 + kc + 32], 0, 1);
  };

  v8f acc0 = {}, acc1 = {};
  loadB(0, 0);
  __syncthreads();

  for (int c = 0; c < chunks; ++c) {
    const int kc  = c << 5;
    const int buf = c & 1;

    // ---- A gather: 16 f16 per lane, contiguous k = 16*lhalf + e ----
    v8u au;
#pragma unroll
    for (int j = 0; j < 8; ++j) {
      int e0  = 2 * j;
      int kk0 = kc + 16 * lhalf + e0;
      float v0 = gatherA(kk0);
      float v1 = gatherA(kk0 + 1);
      au[j] = pack_f16x2(v0, v1);
    }
    v16h a = __builtin_bit_cast(v16h, au);

    // ---- B read from LDS (packed) ----
    v8u b0u, b1u;
    const int kb2 = lhalf * 8;
#pragma unroll
    for (int j = 0; j < 8; ++j) {
      b0u[j] = ldsBp[buf][0][kb2 + j][lm];
      b1u[j] = ldsBp[buf][1][kb2 + j][lm];
    }
    v16h b0 = __builtin_bit_cast(v16h, b0u);
    v16h b1 = __builtin_bit_cast(v16h, b1u);

    acc0 = __builtin_amdgcn_wmma_f32_16x16x32_f16(
        false, a, false, b0, (short)0, acc0, false, false);
    acc1 = __builtin_amdgcn_wmma_f32_16x16x32_f16(
        false, a, false, b1, (short)0, acc1, false, false);

    if (c + 1 < chunks) loadB(kc + 32, buf ^ 1);
    __syncthreads();
  }

  // ---- store: C/D layout lane = N, VGPR r -> M = r (+8 for lanes 16-31) ----
#pragma unroll
  for (int tile = 0; tile < 2; ++tile) {
    const int co = coBase + (tile << 4) + lm;
    if (co < Cout) {
      const float bv = bias[co];
      const v8f& acc = tile ? acc1 : acc0;
#pragma unroll
      for (int r = 0; r < 8; ++r) {
        int mm = mBase + r + lhalf * 8;
        if (mm < Mtot) {
          int ow2 = mm % Wout; int t = mm / Wout;
          int oh2 = t % Hout;  t /= Hout;
          int od2 = t % Dout;  int nb2 = t / Dout;
          y[((nb2 * Cout + co) * Dout + od2) * HWo + oh2 * Wout + ow2] =
              acc[r] + bv;
        }
      }
    }
  }
}

// ===========================================================================
// ConvTranspose3d, k=4, stride=2, pad=1 (PyTorch semantics), decomposed into
// 8 output-parity classes (blockIdx.z) so each class is a dense implicit GEMM
// with K = Cin*8 and a position-independent B matrix. w is (Cin,Cout,4,4,4).
// ===========================================================================
__global__ __launch_bounds__(256) void convT3d_k4s2_wmma(
    const float* __restrict__ x, const float* __restrict__ w,
    const float* __restrict__ bias, float* __restrict__ y,
    int N, int Cin, int Din, int Hin, int Win, int Cout)
{
  __shared__ unsigned ldsBp[2][2][16][16];

  const int Dout = Din * 2, Hout = Hin * 2, Wout = Win * 2;
  const int HWi = Hin * Win, HWo = Hout * Wout;
  const int p  = blockIdx.z;
  const int pd = (p >> 2) & 1, ph = (p >> 1) & 1, pw = p & 1;
  const int Msub = N * Din * HWi;
  const int Ktot = Cin * 8;

  const int lane  = threadIdx.x & 31;
  const int wave  = threadIdx.x >> 5;
  const int lm    = lane & 15;
  const int lhalf = lane >> 4;

  const int coBase = blockIdx.y * 32;
  const int mBase  = (blockIdx.x * 8 + wave) * 16;
  const int m      = mBase + lm;
  const bool mvalid = (m < Msub);

  int qw = 0, qh = 0, qd = 0, nb = 0;
  {
    int mc = mvalid ? m : 0;
    qw = mc % Win; int t = mc / Win;
    qh = t % Hin;  t /= Hin;
    qd = t % Din;  nb = t / Din;
  }
  const int xbase = (nb * Cin) * Din * HWi;
  const int ciStride = Din * HWi;

  const int chunks = (Ktot + 31) >> 5;
  const int kbase8 = lhalf * 8;

  auto gatherA = [&](int kk) -> float {
    int ci = kk >> 3, t = kk & 7;
    int td = (t >> 2) & 1, th = (t >> 1) & 1, tw = t & 1;
    int id = qd + pd - td, ih = qh + ph - th, iw = qw + pw - tw;
    bool valid = mvalid && (kk < Ktot) &&
                 ((unsigned)id < (unsigned)Din) &&
                 ((unsigned)ih < (unsigned)Hin) &&
                 ((unsigned)iw < (unsigned)Win);
    unsigned off =
        valid ? (unsigned)(xbase + ci * ciStride + id * HWi + ih * Win + iw) : 0u;
    float v = x[off];
    return valid ? v : 0.f;
  };

  auto loadB = [&](int kc, int buf) {
#pragma unroll
    for (int i = 0; i < 2; ++i) {
      int e    = threadIdx.x + (i << 8);
      int tile = e >> 8, rem = e & 255;
      int kp   = rem >> 4, n = rem & 15;
      int co   = coBase + (tile << 4) + n;
      bool cv  = (co < Cout);
      float f01[2];   // two consecutive K values for this dword
#pragma unroll
      for (int q = 0; q < 2; ++q) {
        int kg = kc + (kp << 1) + q;
        bool v = cv && (kg < Ktot);
        int ci = kg >> 3, t = kg & 7;
        int kd = (1 - pd) + ((t >> 2) & 1) * 2;
        int kh = (1 - ph) + ((t >> 1) & 1) * 2;
        int kw = (1 - pw) + (t & 1) * 2;
        unsigned off =
            v ? (unsigned)(((ci * Cout + co) * 4 + kd) * 16 + kh * 4 + kw) : 0u;
        float f = w[off];
        f01[q] = v ? f : 0.f;
      }
      ldsBp[buf][tile][kp][n] = pack_f16x2(f01[0], f01[1]);
    }
  };

  v8f acc0 = {}, acc1 = {};
  loadB(0, 0);
  __syncthreads();

  for (int c = 0; c < chunks; ++c) {
    const int kc  = c << 5;
    const int buf = c & 1;

    v8u au;
#pragma unroll
    for (int j = 0; j < 8; ++j) {
      int e0  = 2 * j;
      int kk0 = kc + 16 * lhalf + e0;
      float v0 = gatherA(kk0);
      float v1 = gatherA(kk0 + 1);
      au[j] = pack_f16x2(v0, v1);
    }
    v16h a = __builtin_bit_cast(v16h, au);

    v8u b0u, b1u;
    const int kb2 = lhalf * 8;
#pragma unroll
    for (int j = 0; j < 8; ++j) {
      b0u[j] = ldsBp[buf][0][kb2 + j][lm];
      b1u[j] = ldsBp[buf][1][kb2 + j][lm];
    }
    v16h b0 = __builtin_bit_cast(v16h, b0u);
    v16h b1 = __builtin_bit_cast(v16h, b1u);

    acc0 = __builtin_amdgcn_wmma_f32_16x16x32_f16(
        false, a, false, b0, (short)0, acc0, false, false);
    acc1 = __builtin_amdgcn_wmma_f32_16x16x32_f16(
        false, a, false, b1, (short)0, acc1, false, false);

    if (c + 1 < chunks) loadB(kc + 32, buf ^ 1);
    __syncthreads();
  }

#pragma unroll
  for (int tile = 0; tile < 2; ++tile) {
    const int co = coBase + (tile << 4) + lm;
    if (co < Cout) {
      const float bv = bias[co];
      const v8f& acc = tile ? acc1 : acc0;
#pragma unroll
      for (int r = 0; r < 8; ++r) {
        int mm = mBase + r + lhalf * 8;
        if (mm < Msub) {
          int qw2 = mm % Win; int t = mm / Win;
          int qh2 = t % Hin;  t /= Hin;
          int qd2 = t % Din;  int nb2 = t / Din;
          int od = 2 * qd2 + pd, oh = 2 * qh2 + ph, ow = 2 * qw2 + pw;
          y[((nb2 * Cout + co) * Dout + od) * HWo + oh * Wout + ow] =
              acc[r] + bv;
        }
      }
    }
  }
}

// ===========================================================================
// GroupNorm: stats pass (mean, rstd per (batch, group)) + fused apply/GELU.
// ===========================================================================
__global__ __launch_bounds__(256) void gn_stats(
    const float* __restrict__ x, float* __restrict__ stats, int C, int DHW)
{
  const int bg = blockIdx.x;                    // b * GN_GROUPS + g
  const int cg = C / GN_GROUPS;
  const int n  = cg * DHW;                      // <= 2M elements, fits int
  const float* xb = x + (long long)bg * n;      // groups are contiguous

  float s1 = 0.f, s2 = 0.f;
  for (int i = threadIdx.x; i < n; i += 256) {
    float v = xb[i];
    s1 += v; s2 += v * v;
  }
  __shared__ float sm1[256], sm2[256];
  sm1[threadIdx.x] = s1; sm2[threadIdx.x] = s2;
  __syncthreads();
  for (int o = 128; o > 0; o >>= 1) {
    if (threadIdx.x < o) {
      sm1[threadIdx.x] += sm1[threadIdx.x + o];
      sm2[threadIdx.x] += sm2[threadIdx.x + o];
    }
    __syncthreads();
  }
  if (threadIdx.x == 0) {
    float inv_n = 1.f / (float)n;
    float mean = sm1[0] * inv_n;
    float var  = sm2[0] * inv_n - mean * mean;
    stats[bg * 2 + 0] = mean;
    stats[bg * 2 + 1] = rsqrtf(var + 1e-5f);
  }
}

// grid = (ceil(DHW/256), N*C) ; channel index from blockIdx.y -> scalar math
__global__ __launch_bounds__(256) void gn_apply(
    float* x, const float* __restrict__ stats,
    const float* __restrict__ s, const float* __restrict__ bb,
    int C, int DHW, int do_gelu)
{
  const int nc = blockIdx.y;                    // b*C + c
  const int c  = nc % C;
  const int cg = C / GN_GROUPS;
  const int bg = (nc / C) * GN_GROUPS + c / cg;
  const int i  = blockIdx.x * 256 + threadIdx.x;
  if (i >= DHW) return;
  float* xp = x + (long long)nc * DHW;
  const float mean = stats[bg * 2 + 0], rstd = stats[bg * 2 + 1];
  const float sc = s[c] * rstd;
  const float sh = bb[c] - mean * sc;
  float v = xp[i] * sc + sh;
  if (do_gelu) v = 0.5f * v * (1.f + erff(v * 0.70710678118654752f));
  xp[i] = v;
}

// ===========================================================================
// Physics constraint: per-(b,v,t1,t2) energy reduce, scalar finalize,
// elementwise constrained output.
// ===========================================================================
__global__ __launch_bounds__(256) void phys_ereduce(
    const float* __restrict__ pred, float* __restrict__ e, int seg)
{
  const float* pb = pred + (long long)blockIdx.x * seg;
  float s = 0.f;
  for (int i = threadIdx.x; i < seg; i += 256) s += pb[i];
  __shared__ float sm[256];
  sm[threadIdx.x] = s; __syncthreads();
  for (int o = 128; o > 0; o >>= 1) {
    if (threadIdx.x < o) sm[threadIdx.x] += sm[threadIdx.x + o];
    __syncthreads();
  }
  if (threadIdx.x == 0) e[blockIdx.x] = sm[0];
}

__global__ void phys_final(const float* __restrict__ e, float* __restrict__ out3)
{
  // e laid out as [B=2][V=5][T1=2][T2=4]
  if (threadIdx.x != 0 || blockIdx.x != 0) return;
  float ev = 0.f, mv = 0.f;
  for (int b = 0; b < 2; ++b)
    for (int v = 0; v < 5; ++v)
      for (int t1 = 0; t1 < 2; ++t1) {
        const float* e4 = e + ((b * 5 + v) * 2 + t1) * 4;
        float s = e4[0] + e4[1] + e4[2] + e4[3];
        float mn = s * 0.25f;
        for (int t2 = 0; t2 < 4; ++t2) ev += fabsf(e4[t2] - mn);
      }
  ev *= (1.f / 80.f);
  for (int b = 0; b < 2; ++b)
    for (int v = 0; v < 5; ++v) {
      float m0 = 0.f, m1 = 0.f;
      for (int t2 = 0; t2 < 4; ++t2) {
        m0 += e[((b * 5 + v) * 2 + 0) * 4 + t2];
        m1 += e[((b * 5 + v) * 2 + 1) * 4 + t2];
      }
      float mn = 0.5f * (m0 + m1);
      mv += fabsf(m0 - mn) + fabsf(m1 - mn);
    }
  mv *= (1.f / 20.f);
  out3[0] = ev; out3[1] = mv; out3[2] = ev + mv;
}

// grid = (524288/256, 10) ; v index from blockIdx.y -> scalar math
__global__ __launch_bounds__(256) void phys_constrain(
    const float* __restrict__ pred, const float* __restrict__ pwv,
    const float* __restrict__ pbv, float* __restrict__ outc)
{
  const int bv = blockIdx.y;                    // b*V + v
  const int v  = bv % 5;
  const long long base = (long long)bv * 524288; // T1*T2*L*H*W
  const int i = blockIdx.x * 256 + threadIdx.x;
  outc[base + i] = pred[base + i] * pwv[v] + pbv[v];
}

// ===========================================================================
// Host orchestration
// ===========================================================================
extern "C" void kernel_launch(void* const* d_in, const int* in_sizes, int n_in,
                              void* d_out, int out_size, void* d_ws, size_t ws_size,
                              hipStream_t stream) {
  (void)in_sizes; (void)n_in; (void)out_size; (void)ws_size;

  // ---- input enumeration: setup_inputs() insertion order ----
  int ii = 0;
  auto F = [&](void) { return (const float*)d_in[ii++]; };
  const float* X    = F();
  const float* in_w = F();
  const float* in_b = F();
  const float* enc_p[4][8];
  for (int i = 0; i < 4; ++i) for (int j = 0; j < 8; ++j) enc_p[i][j] = F();
  const float* attn_p[2][2];
  for (int i = 0; i < 2; ++i) for (int j = 0; j < 2; ++j) attn_p[i][j] = F();
  const float* bot_p[8];
  for (int j = 0; j < 8; ++j) bot_p[j] = F();
  const float* dec_p[4][8];
  for (int i = 0; i < 4; ++i) for (int j = 0; j < 8; ++j) dec_p[i][j] = F();
  const float* out_w  = F();
  const float* out_b  = F();
  const float* phys_w = F();
  const float* phys_b = F();

  // ---- workspace: two ping-pong activation buffers + stats + e[] ----
  const long long BUF = 33554432LL;            // 2*32*32*128*128 floats (max act)
  float* bufA  = (float*)d_ws;
  float* bufB  = bufA + BUF;
  float* stats = bufA + 2 * BUF;               // up to 16 (mean,rstd) pairs
  float* ebuf  = stats + 64;                   // e[80]

  auto conv = [&](const float* xin, const float* wgt, const float* bs, float* yout,
                  int Cin, int Di, int Hi, int Wi, int Cout, int st) {
    int Do = (Di - 1) / st + 1, Ho = (Hi - 1) / st + 1, Wo = (Wi - 1) / st + 1;
    int Mtot = 2 * Do * Ho * Wo;
    dim3 g((Mtot + 127) / 128, (Cout + 31) / 32, 1);
    conv3d_k3_wmma<<<g, 256, 0, stream>>>(xin, wgt, bs, yout,
                                          2, Cin, Di, Hi, Wi, Cout, Do, Ho, Wo, st);
  };
  auto convT = [&](const float* xin, const float* wgt, const float* bs, float* yout,
                   int Cin, int Di, int Hi, int Wi, int Cout) {
    int Msub = 2 * Di * Hi * Wi;
    dim3 g((Msub + 127) / 128, (Cout + 31) / 32, 8);
    convT3d_k4s2_wmma<<<g, 256, 0, stream>>>(xin, wgt, bs, yout,
                                             2, Cin, Di, Hi, Wi, Cout);
  };
  auto gnorm = [&](float* xio, const float* s, const float* b,
                   int C, int DHW, int gelu) {
    gn_stats<<<2 * GN_GROUPS, 256, 0, stream>>>(xio, stats, C, DHW);
    dim3 g((DHW + 255) / 256, 2 * C, 1);
    gn_apply<<<g, 256, 0, stream>>>(xio, stats, s, b, C, DHW, gelu);
  };

  // ---- stem: x viewed as (2,5,32,128,128) -> (2,32,32,128,128) ----
  conv(X, in_w, in_b, bufA, 5, 32, 128, 128, 32, 1);

  // ---- encoder ----
  int C = 32, D = 32, H = 128, W = 128;
  for (int i = 0; i < 4; ++i) {
    int Co = 2 * C, Do = D / 2, Ho = H / 2, Wo = W / 2;
    conv(bufA, enc_p[i][0], enc_p[i][1], bufB, C, D, H, W, Co, 2);
    gnorm(bufB, enc_p[i][2], enc_p[i][3], Co, Do * Ho * Wo, 1);
    conv(bufB, enc_p[i][4], enc_p[i][5], bufA, Co, Do, Ho, Wo, Co, 1);
    gnorm(bufA, enc_p[i][6], enc_p[i][7], Co, Do * Ho * Wo, 1);
    if (i & 1) {
      int a = i >> 1;  // attention modules only apply their GroupNorm
      gnorm(bufA, attn_p[a][0], attn_p[a][1], Co, Do * Ho * Wo, 0);
    }
    C = Co; D = Do; H = Ho; W = Wo;
  }

  // ---- bottleneck: 512 -> 1024 -> 512 at (2,8,8) ----
  conv(bufA, bot_p[0], bot_p[1], bufB, C, D, H, W, 2 * C, 1);
  gnorm(bufB, bot_p[2], bot_p[3], 2 * C, D * H * W, 1);
  conv(bufB, bot_p[4], bot_p[5], bufA, 2 * C, D, H, W, C, 1);
  gnorm(bufA, bot_p[6], bot_p[7], C, D * H * W, 1);

  // ---- decoder ----
  for (int i = 0; i < 4; ++i) {
    int Co = C / 2, Do = D * 2, Ho = H * 2, Wo = W * 2;
    convT(bufA, dec_p[i][0], dec_p[i][1], bufB, C, D, H, W, Co);
    gnorm(bufB, dec_p[i][2], dec_p[i][3], Co, Do * Ho * Wo, 1);
    conv(bufB, dec_p[i][4], dec_p[i][5], bufA, Co, Do, Ho, Wo, Co, 1);
    gnorm(bufA, dec_p[i][6], dec_p[i][7], Co, Do * Ho * Wo, 1);
    C = Co; D = Do; H = Ho; W = Wo;
  }

  // ---- output conv straight into d_out (pred region) ----
  float* pred = (float*)d_out;
  conv(bufA, out_w, out_b, pred, 32, 32, 128, 128, 5, 1);

  // ---- physics constraint ----
  const long long P = 5242880LL;               // 2*5*2*4*4*128*128
  phys_ereduce<<<80, 256, 0, stream>>>(pred, ebuf, 4 * 128 * 128);
  phys_final<<<1, 32, 0, stream>>>(ebuf, pred + 2 * P);
  phys_constrain<<<dim3(2048, 10, 1), 256, 0, stream>>>(
      pred, phys_w, phys_b, pred + P);
}
